// TransformerEncoder_77489799954958
// MI455X (gfx1250) — matrix operand order repaired and checked by reading, not compiled, not run on hardware
//
#include <hip/hip_runtime.h>
#include <hip/hip_bf16.h>
#include <math.h>

// ---------------------------------------------------------------------------
// TransformerEncoder layer for MI455X (gfx1250, wave32, WMMA bf16 path).
// B=8, S=1024, D=1024, H=16, DH=64. All GEMMs via v_wmma_f32_16x16x32_bf16.
// ---------------------------------------------------------------------------

#define BB   8
#define SS   1024
#define DD   1024
#define HH   16
#define DHH  64
#define MM   (BB * SS)          // 8192 rows
#define D2   (2 * DD)           // 2048

typedef __attribute__((ext_vector_type(16))) __bf16 v16bf;
typedef __attribute__((ext_vector_type(8)))  __bf16 v8bf;
typedef __attribute__((ext_vector_type(8)))  float  v8f;

// ---------------- WMMA helpers (contiguous k = 16h + s fragments for A and B) -----

__device__ __forceinline__ v8f wmma_bf16(v16bf a, v16bf b, v8f c) {
  // (neg_a, A, neg_b, B, c_mod, C, reuse_a, reuse_b)
  return __builtin_amdgcn_wmma_f32_16x16x32_bf16(false, a, false, b,
                                                 (short)0, c, false, false);
}

// A fragment 16x32 bf16, row-major source with leading dim `ld`.
// lanes 0-15: M=lane, K = k0..k0+15
// lanes16-31: M=lane-16, K = k0+16..k0+31
__device__ __forceinline__ v16bf load_a_frag(const __bf16* __restrict__ base,
                                             int ld, int row, int k0, int laneHi) {
  const __bf16* p = base + (size_t)row * ld + k0 + (laneHi ? 16 : 0);
  v8bf lo = *(const v8bf*)(p);
  v8bf hi = *(const v8bf*)(p + 8);
  v16bf a;
#pragma unroll
  for (int i = 0; i < 8; ++i) { a[i] = lo[i]; a[8 + i] = hi[i]; }
  return a;
}

// B fragment 32x16 bf16 from a K-contiguous column (Bt[n][k] layout).
// lanes 0-15: col=lane, K=k0..k0+15 ; lanes 16-31: col=lane-16, K=k0+16..k0+31
__device__ __forceinline__ v16bf load_b_frag(const __bf16* __restrict__ colbase,
                                             int k0, int laneHi) {
  return *(const v16bf*)(colbase + k0 + (laneHi ? 16 : 0));
}

// ---------------- weight transpose + fp32->bf16 conversion -----------------
// out[b][n][k] = bf16(in[b][k][n]);  in: [batch][R][C], out: [batch][C][R]
__global__ __launch_bounds__(256) void k_transpose_bf16(
    const float* __restrict__ in, __bf16* __restrict__ out,
    int R, int C, long total) {
  long idx = (long)blockIdx.x * 256 + threadIdx.x;
  if (idx >= total) return;
  long k = idx % R;
  long t = idx / R;
  long n = t % C;
  long b = t / C;
  out[idx] = (__bf16)in[(b * R + k) * C + n];
}

// ---------------- fused LayerNorm (+residual, +GELU, multi-output) ---------
__global__ __launch_bounds__(256) void k_ln_fused(
    const float* __restrict__ in1, const float* __restrict__ addin,
    const float* __restrict__ g, const float* __restrict__ bvec,
    float* __restrict__ xout,        // optional: residual sum (pre-LN input)
    __bf16* __restrict__ ybf,        // optional: bf16 normalized (+gelu) out
    float* __restrict__ yf,          // optional: f32 out = y (+ postres)
    const float* __restrict__ postres,
    int W, int do_gelu) {
  size_t base = (size_t)blockIdx.x * W;
  int tid = threadIdx.x, lane = tid & 31, wave = tid >> 5;
  __shared__ float rs[8], rq[8];
  float s = 0.f, q = 0.f;
  for (int i = tid; i < W; i += 256) {
    float x = in1[base + i];
    if (addin) x += addin[base + i];
    if (xout)  xout[base + i] = x;
    s += x; q += x * x;
  }
#pragma unroll
  for (int off = 16; off >= 1; off >>= 1) {
    s += __shfl_xor(s, off);
    q += __shfl_xor(q, off);
  }
  if (lane == 0) { rs[wave] = s; rq[wave] = q; }
  __syncthreads();
  if (tid == 0) {
    float ts = 0.f, tq = 0.f;
    for (int w = 0; w < 8; ++w) { ts += rs[w]; tq += rq[w]; }
    rs[0] = ts; rq[0] = tq;
  }
  __syncthreads();
  float mean = rs[0] / (float)W;
  float var  = rq[0] / (float)W - mean * mean;
  float rstd = rsqrtf(var + 1e-5f);
  for (int i = tid; i < W; i += 256) {
    float x = in1[base + i];
    if (addin) x += addin[base + i];
    float y = (x - mean) * rstd * g[i] + bvec[i];
    if (do_gelu) y = 0.5f * y * (1.f + erff(y * 0.70710678118654752f));
    if (ybf) ybf[base + i] = (__bf16)y;
    if (yf)  yf[base + i] = y + (postres ? postres[base + i] : 0.f);
  }
}

// ---------------- generic big-tile GEMM: C = A(bf16) x Bt(bf16)^T + bias ---
// grid: (N/128, M/128), block 256 = 8 waves in 4(M) x 2(N) grid.
// Wave tile 32x64 (2x4 WMMA frags): 8 WMMA per K-step vs 12 b128 loads.
// Bt: [N][K] row-major bf16 (pre-transposed), K = KD (compile-time).
// MODE 0: C fp32 [M][N].
// MODE 1: bf16 per-head scatter  dst[((b*H+h)*S+s)*DH + c]   (q, k)
// MODE 2: bf16 per-head transp.  dst[((b*H+h)*DH+c)*S + s]   (v)
template <int KD, int MODE>
__global__ __launch_bounds__(256) void k_gemm(
    const __bf16* __restrict__ A, const __bf16* __restrict__ Bt,
    const float* __restrict__ bias,
    float* __restrict__ Cf, __bf16* __restrict__ Cb, int Ndim) {
  int lane = threadIdx.x & 31, wave = threadIdx.x >> 5;
  int wm = wave >> 1, wn = wave & 1;
  int bm = blockIdx.y * 128 + wm * 32;
  int bn = blockIdx.x * 128 + wn * 64;
  int laneHi = lane >> 4, lan = lane & 15;

  const __bf16* arow0 = A + (size_t)(bm + lan) * KD;
  const __bf16* brow0 = Bt + (size_t)(bn + lan) * KD;

  v8f c[2][4] = {};
#pragma unroll 4
  for (int k0 = 0; k0 < KD; k0 += 32) {
    v16bf a0 = load_a_frag(arow0, KD, 0,  k0, laneHi);
    v16bf a1 = load_a_frag(arow0, KD, 16, k0, laneHi);
    v16bf b0 = load_b_frag(brow0,               k0, laneHi);
    v16bf b1 = load_b_frag(brow0 + 16 * KD,     k0, laneHi);
    v16bf b2 = load_b_frag(brow0 + 32 * KD,     k0, laneHi);
    v16bf b3 = load_b_frag(brow0 + 48 * KD,     k0, laneHi);
    c[0][0] = wmma_bf16(a0, b0, c[0][0]);
    c[0][1] = wmma_bf16(a0, b1, c[0][1]);
    c[0][2] = wmma_bf16(a0, b2, c[0][2]);
    c[0][3] = wmma_bf16(a0, b3, c[0][3]);
    c[1][0] = wmma_bf16(a1, b0, c[1][0]);
    c[1][1] = wmma_bf16(a1, b1, c[1][1]);
    c[1][2] = wmma_bf16(a1, b2, c[1][2]);
    c[1][3] = wmma_bf16(a1, b3, c[1][3]);
  }
#pragma unroll
  for (int mt = 0; mt < 2; ++mt) {
#pragma unroll
    for (int nt = 0; nt < 4; ++nt) {
#pragma unroll
      for (int i = 0; i < 8; ++i) {
        int row = bm + mt * 16 + i + laneHi * 8;   // global M row
        int col = bn + nt * 16 + lan;              // global N col
        float v = c[mt][nt][i] + bias[col];
        if (MODE == 0) {
          Cf[(size_t)row * Ndim + col] = v;
        } else {
          int bidx = row >> 10;                    // row / S
          int sidx = row & (SS - 1);               // row % S
          int h    = col >> 6;                     // col / DH
          int cc   = col & (DHH - 1);              // col % DH
          size_t bh = (size_t)(bidx * HH + h);
          if (MODE == 1) Cb[(bh * SS + sidx) * DHH + cc] = (__bf16)v;
          else           Cb[(bh * DHH + cc) * SS + sidx] = (__bf16)v;
        }
      }
    }
  }
}

// ---------------- flash attention ------------------------------------------
// grid: (SS/128, BB*HH), block 256 (8 waves). Each wave: 16 query rows,
// streams keys in blocks of 32, online softmax, O accum 16x64 f32 in regs.
__global__ __launch_bounds__(256) void k_attn(
    const __bf16* __restrict__ q,   // [B,H,S,DH]
    const __bf16* __restrict__ k,   // [B,H,S,DH]
    const __bf16* __restrict__ vT,  // [B,H,DH,S]
    float* __restrict__ o) {        // [B,S,D] (heads concatenated)
  const float SCL = 0.125f;         // 1/sqrt(DH)
  int bh = blockIdx.y;
  int bidx = bh / HH, h = bh % HH;
  int lane = threadIdx.x & 31, wave = threadIdx.x >> 5;
  int laneHi = lane >> 4, lan = lane & 15;
  int qrow0 = blockIdx.x * 128 + wave * 16;

  const __bf16* qh = q  + (size_t)bh * SS * DHH;
  const __bf16* kh = k  + (size_t)bh * SS * DHH;
  const __bf16* vh = vT + (size_t)bh * DHH * SS;

  __shared__ __bf16 Pl[8][16][32];  // per-wave P relayout buffer (8 KB)

  // Q fragments for both K(dh)-chunks, resident all loop
  v16bf aq0 = load_a_frag(qh, DHH, qrow0 + lan, 0,  laneHi);
  v16bf aq1 = load_a_frag(qh, DHH, qrow0 + lan, 32, laneHi);

  v8f ofr[4] = {v8f{}, v8f{}, v8f{}, v8f{}};
  float m[8], l[8];
#pragma unroll
  for (int i = 0; i < 8; ++i) { m[i] = -INFINITY; l[i] = 0.f; }

  for (int kb = 0; kb < SS; kb += 32) {
    // scores tile 16x32 = two 16x16 frags, K(dh)=64 in two chunks
    v8f s0 = {}, s1 = {};
    {
      const __bf16* kp00 = kh + (size_t)(kb + lan) * DHH      + (laneHi ? 16 : 0);
      const __bf16* kp10 = kh + (size_t)(kb + 16 + lan) * DHH + (laneHi ? 16 : 0);
      v16bf bk00 = *(const v16bf*)(kp00);
      v16bf bk01 = *(const v16bf*)(kp00 + 32);
      v16bf bk10 = *(const v16bf*)(kp10);
      v16bf bk11 = *(const v16bf*)(kp10 + 32);
      s0 = wmma_bf16(aq0, bk00, s0);
      s0 = wmma_bf16(aq1, bk01, s0);
      s1 = wmma_bf16(aq0, bk10, s1);
      s1 = wmma_bf16(aq1, bk11, s1);
    }
    // online softmax per row (rows i / i+8 by lane half)
#pragma unroll
    for (int i = 0; i < 8; ++i) {
      float x0 = s0[i] * SCL, x1 = s1[i] * SCL;
      float rmax = fmaxf(x0, x1);
#pragma unroll
      for (int off = 8; off >= 1; off >>= 1)
        rmax = fmaxf(rmax, __shfl_xor(rmax, off));
      float mnew  = fmaxf(m[i], rmax);
      float alpha = __expf(m[i] - mnew);
      float p0 = __expf(x0 - mnew);
      float p1 = __expf(x1 - mnew);
      float rsum = p0 + p1;
#pragma unroll
      for (int off = 8; off >= 1; off >>= 1) rsum += __shfl_xor(rsum, off);
      l[i] = l[i] * alpha + rsum;
      m[i] = mnew;
#pragma unroll
      for (int f = 0; f < 4; ++f) ofr[f][i] *= alpha;
      int r = i + laneHi * 8;
      Pl[wave][r][lan]      = (__bf16)p0;
      Pl[wave][r][16 + lan] = (__bf16)p1;
    }
    // reload P as A fragment (wave-private LDS, DS in-order, no barrier)
    const __bf16* pr = &Pl[wave][lan][0];
    v8bf plo = *(const v8bf*)(pr + (laneHi ? 16 : 0));
    v8bf phi = *(const v8bf*)(pr + 8 + (laneHi ? 16 : 0));
    v16bf ap;
#pragma unroll
    for (int i = 0; i < 8; ++i) { ap[i] = plo[i]; ap[8 + i] = phi[i]; }
    // O += P x V  (V read K-contiguous from vT[dh][s])
#pragma unroll
    for (int f = 0; f < 4; ++f) {
      const __bf16* vp = vh + (size_t)(f * 16 + lan) * SS + kb + (laneHi ? 16 : 0);
      v16bf bv = *(const v16bf*)vp;
      ofr[f] = wmma_bf16(ap, bv, ofr[f]);
    }
  }
  // finalize: O /= l, write concat-head layout [B,S,D]
#pragma unroll
  for (int i = 0; i < 8; ++i) {
    float inv = 1.f / l[i];
    int r = qrow0 + i + laneHi * 8;
#pragma unroll
    for (int f = 0; f < 4; ++f) {
      o[((size_t)bidx * SS + r) * DD + h * DHH + f * 16 + lan] = ofr[f][i] * inv;
    }
  }
}

// ---------------------------------------------------------------------------
extern "C" void kernel_launch(void* const* d_in, const int* in_sizes, int n_in,
                              void* d_out, int out_size, void* d_ws, size_t ws_size,
                              hipStream_t stream) {
  (void)in_sizes; (void)n_in; (void)out_size; (void)ws_size;
  const float* X      = (const float*)d_in[0];
  const float* Wq     = (const float*)d_in[1];
  const float* bq     = (const float*)d_in[2];
  const float* Wk     = (const float*)d_in[3];
  const float* bk     = (const float*)d_in[4];
  const float* Wv     = (const float*)d_in[5];
  const float* bv     = (const float*)d_in[6];
  const float* pre_g  = (const float*)d_in[7];
  const float* pre_b  = (const float*)d_in[8];
  const float* post_g = (const float*)d_in[9];
  const float* post_b = (const float*)d_in[10];
  const float* W1     = (const float*)d_in[11];
  const float* b1     = (const float*)d_in[12];
  const float* ln1_g  = (const float*)d_in[13];
  const float* ln1_b  = (const float*)d_in[14];
  const float* W2     = (const float*)d_in[15];
  const float* b2     = (const float*)d_in[16];
  const float* ln2_g  = (const float*)d_in[17];
  const float* ln2_b  = (const float*)d_in[18];
  float* out = (float*)d_out;

  char* ws = (char*)d_ws;
  const size_t MB = (size_t)1 << 20;
  // persistent bf16 weights (all pre-transposed to Bt[N][K])
  __bf16* Wqt = (__bf16*)(ws + 0 * MB);    // [1024,1024] (=[H,64,1024]) 2 MB
  __bf16* Wkt = (__bf16*)(ws + 2 * MB);    //                            2 MB
  __bf16* Wvt = (__bf16*)(ws + 4 * MB);    //                            2 MB
  __bf16* W1t = (__bf16*)(ws + 6 * MB);    // [2048,1024]                4 MB
  __bf16* W2t = (__bf16*)(ws + 10 * MB);   // [1024,2048]                4 MB
  // poolA [14,78) MB — lifetime-aliased
  __bf16* tbf   = (__bf16*)(ws + 14 * MB); // LN(X) bf16 [8192,1024]   16 MB
  __bf16* qbf   = (__bf16*)(ws + 30 * MB); // [B,H,S,DH]               16 MB
  __bf16* kbf   = (__bf16*)(ws + 46 * MB); // [B,H,S,DH]               16 MB
  __bf16* vTbf  = (__bf16*)(ws + 62 * MB); // [B,H,DH,S]               16 MB
  float*  h1pre = (float*)(ws + 14 * MB);  // [8192,2048] f32, reuses t/q/k/vT
  float*  h2pre = (float*)(ws + 14 * MB);  // [8192,1024] f32, reuses h1pre
  // poolB [78,110) MB
  float*  obuf  = (float*)(ws + 78 * MB);  // attn out [8192,1024] f32  32 MB
  __bf16* h1bf  = (__bf16*)(ws + 78 * MB); // gelu(LN(h1pre)) bf16, reuses obuf
  // persistent late-stage
  float*  X2    = (float*)(ws + 110 * MB); // X + attn  f32             32 MB
  __bf16* t2bf  = (__bf16*)(ws + 142 * MB);// LN(X2) bf16               16 MB

  // 1. weight transpose + bf16 convert
  {
    long tq = (long)HH * DD * DHH;               // 1,048,576
    k_transpose_bf16<<<(tq + 255) / 256, 256, 0, stream>>>(Wq, Wqt, DD, DHH, tq);
    k_transpose_bf16<<<(tq + 255) / 256, 256, 0, stream>>>(Wk, Wkt, DD, DHH, tq);
    k_transpose_bf16<<<(tq + 255) / 256, 256, 0, stream>>>(Wv, Wvt, DD, DHH, tq);
    long t1 = (long)DD * D2;                     // 2,097,152
    k_transpose_bf16<<<(t1 + 255) / 256, 256, 0, stream>>>(W1, W1t, DD, D2, t1);
    k_transpose_bf16<<<(t1 + 255) / 256, 256, 0, stream>>>(W2, W2t, D2, DD, t1);
  }
  // 2. pre-LN: tbf = bf16(LN(X))
  k_ln_fused<<<MM, 256, 0, stream>>>(X, nullptr, pre_g, pre_b,
                                     nullptr, tbf, nullptr, nullptr, DD, 0);
  // 3. QKV projections (WMMA big-tile GEMM, N folded over heads = 1024)
  {
    dim3 g(DD / 128, MM / 128);
    k_gemm<DD, 1><<<g, 256, 0, stream>>>(tbf, Wqt, bq, nullptr, qbf,  DD);
    k_gemm<DD, 1><<<g, 256, 0, stream>>>(tbf, Wkt, bk, nullptr, kbf,  DD);
    k_gemm<DD, 2><<<g, 256, 0, stream>>>(tbf, Wvt, bv, nullptr, vTbf, DD);
  }
  // 4. flash attention (WMMA)
  {
    dim3 g(SS / 128, BB * HH);
    k_attn<<<g, 256, 0, stream>>>(qbf, kbf, vTbf, obuf);
  }
  // 5. residual + post-LN: X2 = X + o ; t2 = bf16(LN(X2))
  k_ln_fused<<<MM, 256, 0, stream>>>(obuf, X, post_g, post_b,
                                     X2, t2bf, nullptr, nullptr, DD, 0);
  // 6. h1pre = t2 @ W1 + b1   (M=8192, N=2048, K=1024)
  {
    dim3 g(D2 / 128, MM / 128);
    k_gemm<DD, 0><<<g, 256, 0, stream>>>(t2bf, W1t, b1, h1pre, nullptr, D2);
  }
  // 7. h1 = bf16(gelu(LN(h1pre)))
  k_ln_fused<<<MM, 256, 0, stream>>>(h1pre, nullptr, ln1_g, ln1_b,
                                     nullptr, h1bf, nullptr, nullptr, D2, 1);
  // 8. h2pre = h1 @ W2 + b2   (M=8192, N=1024, K=2048)
  {
    dim3 g(DD / 128, MM / 128);
    k_gemm<D2, 0><<<g, 256, 0, stream>>>(h1bf, W2t, b2, h2pre, nullptr, DD);
  }
  // 9. out = X2 + gelu(LN(h2pre))
  k_ln_fused<<<MM, 256, 0, stream>>>(h2pre, nullptr, ln2_g, ln2_b,
                                     nullptr, nullptr, out, X2, DD, 1);
}
